// RNN_3358664425663
// MI455X (gfx1250) — matrix operand order repaired and checked by reading, not compiled, not run on hardware
//
#include <hip/hip_runtime.h>

#define HID   128
#define TLEN  512
#define BATCH 1024
#define ROWS  16          // batch rows per workgroup
#define NTHREADS 256      // 8 waves (wave32)

// 2*log2(e): folded into W_ih / bias / W_hh so the WMMA produces the
// pre-activation already scaled for exp2 (no per-step v_pk_mul).
#define TWO_LOG2E 2.8853900817779268f

typedef __attribute__((ext_vector_type(16))) _Float16 v16h;
typedef __attribute__((ext_vector_type(8)))  float    v8f;
typedef __attribute__((ext_vector_type(4)))  float    f32x4;

union AFrag { v16h v; f32x4 q[2]; };

__device__ __forceinline__ float fast_tanh_pre_scaled(float z) {
    // z is already 2*log2(e)*x. tanh(x) = 1 - 2/(2^z + 1). No clamp needed:
    //   2^z = +inf -> rcp = 0 -> 1 ;  2^z -> 0 -> rcp(1) = 1 -> -1.
    float e = __builtin_amdgcn_exp2f(z);           // v_exp_f32 (TRANS)
    float r = __builtin_amdgcn_rcpf(e + 1.0f);     // v_rcp_f32 (TRANS)
    return fmaf(-2.0f, r, 1.0f);                   // fuses into v_fma_mixlo/hi_f16
}

// One scan step: read h from `hrow`'s buffer, write tanh(pre-act) into `hw`'s
// buffer, and recompute xpacc for the NEXT step (from xnext) in the WMMA shadow.
__device__ __forceinline__ void rnn_step(const _Float16* __restrict__ hrow,
                                         _Float16* __restrict__ hw,
                                         const float* __restrict__ xnext,
                                         const float wih, const float bias,
                                         const v16h* __restrict__ bfrag,
                                         v8f& xpacc)
{
    // A fragments: row M = lane&15; per-lane K chunks at 8*halfw and 8*halfw+16
    AFrag a[4];
    #pragma unroll
    for (int ks = 0; ks < 4; ++ks) {
        const _Float16* ap = hrow + 32 * ks;
        a[ks].q[0] = *(const f32x4*)(ap);        // ds_load_b128
        a[ks].q[1] = *(const f32x4*)(ap + 16);   // ds_load_b128
    }

    // Two independent accumulation chains -> halves serialized XDL latency.
    v8f acc0 = __builtin_amdgcn_wmma_f32_16x16x32_f16(
                   false, a[0].v, false, bfrag[0], (short)0, xpacc, false, false);
    v8f acc1;
    #pragma unroll
    for (int r = 0; r < 8; ++r) acc1[r] = 0.0f;
    acc1 = __builtin_amdgcn_wmma_f32_16x16x32_f16(
               false, a[1].v, false, bfrag[1], (short)0, acc1, false, false);
    acc0 = __builtin_amdgcn_wmma_f32_16x16x32_f16(
               false, a[2].v, false, bfrag[2], (short)0, acc0, false, false);
    acc1 = __builtin_amdgcn_wmma_f32_16x16x32_f16(
               false, a[3].v, false, bfrag[3], (short)0, acc1, false, false);

    // xp for the next step (lds_x is read-only; independent of barrier).
    f32x4 x0 = *(const f32x4*)(xnext);
    f32x4 x1 = *(const f32x4*)(xnext + 4);
    xpacc[0] = fmaf(x0[0], wih, bias);
    xpacc[1] = fmaf(x0[1], wih, bias);
    xpacc[2] = fmaf(x0[2], wih, bias);
    xpacc[3] = fmaf(x0[3], wih, bias);
    xpacc[4] = fmaf(x1[0], wih, bias);
    xpacc[5] = fmaf(x1[1], wih, bias);
    xpacc[6] = fmaf(x1[2], wih, bias);
    xpacc[7] = fmaf(x1[3], wih, bias);

    // activation + writeback: lane writes h[r+8*halfw][n] of the other buffer
    #pragma unroll
    for (int r = 0; r < 8; ++r)
        hw[r * HID] = (_Float16)fast_tanh_pre_scaled(acc0[r] + acc1[r]);

    __syncthreads();
}

__global__ __launch_bounds__(NTHREADS, 1)
void rnn_scan_wmma(const float* __restrict__ x,      // [B, T]
                   const float* __restrict__ W_ih,   // [H]
                   const float* __restrict__ W_hh,   // [H, H] row-major
                   const float* __restrict__ b_ih,   // [H]
                   const float* __restrict__ b_hh,   // [H]
                   const float* __restrict__ W_out,  // [H]
                   const float* __restrict__ b_out,  // [1]
                   float* __restrict__ y)            // [B]
{
    __shared__ float    lds_x[TLEN * ROWS];        // 32 KB, x transposed: [t][m]
    __shared__ _Float16 lds_h[2][ROWS * HID];      // 8 KB, double-buffered h (f16)

    const int tid   = threadIdx.x;
    const int wave  = tid >> 5;          // 0..7 -> N-tile (16 columns each)
    const int lane  = tid & 31;
    const int halfw = lane >> 4;         // 0 | 1
    const int l16   = lane & 15;
    const int n     = wave * 16 + l16;   // this lane's output column, 0..127
    const int b0    = blockIdx.x * ROWS;

    // ---- per-lane constants (rank-1 input projection), pre-scaled by 2*log2e ----
    const float wih  = W_ih[n] * TWO_LOG2E;
    const float bias = (b_ih[n] + b_hh[n]) * TWO_LOG2E;

    // ---- stationary B fragments: B[k][n] = W_hh[n][k] * 2*log2e ----
    // B layout matching A: lane half h holds K=8h..8h+7 (elems 0-7) and
    // K=16+8h..16+8h+7 (elems 8-15) of column N=lane&15, from one row of W_hh.
    v16h bfrag[4];
    #pragma unroll
    for (int ks = 0; ks < 4; ++ks) {
        const int kbase = 32 * ks + 8 * halfw;
        const float* src = W_hh + n * HID + kbase;   // two 8-element runs, 16B-aligned
        v16h b;
        #pragma unroll
        for (int j = 0; j < 16; ++j) b[j] = (_Float16)(src[j + (j >= 8 ? 8 : 0)] * TWO_LOG2E);
        bfrag[ks] = b;
    }

    // ---- stage x tile transposed into LDS (coalesced on t) ----
    for (int idx = tid; idx < ROWS * TLEN; idx += NTHREADS) {
        int m = idx >> 9;             // idx / TLEN
        int t = idx & (TLEN - 1);
        lds_x[t * ROWS + m] = x[(size_t)(b0 + m) * TLEN + t];
    }
    // h0 = 0
    for (int idx = tid; idx < ROWS * HID; idx += NTHREADS)
        lds_h[0][idx] = (_Float16)0.0f;
    __syncthreads();

    // ---- loop-invariant ping/pong pointers (no per-step address ALU) ----
    const _Float16* hrow0 = &lds_h[0][l16 * HID + 8 * halfw];  // read buf0
    const _Float16* hrow1 = &lds_h[1][l16 * HID + 8 * halfw];  // read buf1
    _Float16* hw0 = &lds_h[1][8 * halfw * HID + n];            // even step writes buf1
    _Float16* hw1 = &lds_h[0][8 * halfw * HID + n];            // odd step writes buf0
    const float* xbase = &lds_x[halfw * 8];

    // ---- software-pipelined xp: compute xp(0) before entering the scan ----
    v8f xpacc;
    {
        f32x4 x0 = *(const f32x4*)(xbase);
        f32x4 x1 = *(const f32x4*)(xbase + 4);
        xpacc[0] = fmaf(x0[0], wih, bias);
        xpacc[1] = fmaf(x0[1], wih, bias);
        xpacc[2] = fmaf(x0[2], wih, bias);
        xpacc[3] = fmaf(x0[3], wih, bias);
        xpacc[4] = fmaf(x1[0], wih, bias);
        xpacc[5] = fmaf(x1[1], wih, bias);
        xpacc[6] = fmaf(x1[2], wih, bias);
        xpacc[7] = fmaf(x1[3], wih, bias);
    }

    // ---- sequential scan over T, unrolled by 2 (fixed buffers per half) ----
    for (int t = 0; t < TLEN; t += 2) {
        // even step: buf0 -> buf1 ; prefetch xp(t+1)
        rnn_step(hrow0, hw0, xbase + (t + 1) * ROWS, wih, bias, bfrag, xpacc);
        // odd step: buf1 -> buf0 ; prefetch xp(t+2) (clamped on last iteration)
        const int tn = (t + 2 < TLEN) ? (t + 2) : (TLEN - 1);
        rnn_step(hrow1, hw1, xbase + tn * ROWS, wih, bias, bfrag, xpacc);
    }

    // ---- output projection y = h_last @ W_out^T + b_out (h_last in buf0) ----
    if (tid < ROWS) {
        const _Float16* hf = &lds_h[0][tid * HID];
        float s = b_out[0];
        #pragma unroll 8
        for (int k = 0; k < HID; ++k)
            s = fmaf((float)hf[k], W_out[k], s);
        y[b0 + tid] = s;
    }
}

extern "C" void kernel_launch(void* const* d_in, const int* in_sizes, int n_in,
                              void* d_out, int out_size, void* d_ws, size_t ws_size,
                              hipStream_t stream) {
    const float* x     = (const float*)d_in[0];
    const float* W_ih  = (const float*)d_in[1];
    const float* W_hh  = (const float*)d_in[2];
    const float* b_ih  = (const float*)d_in[3];
    const float* b_hh  = (const float*)d_in[4];
    const float* W_out = (const float*)d_in[5];
    const float* b_out = (const float*)d_in[6];
    float* y = (float*)d_out;

    rnn_scan_wmma<<<BATCH / ROWS, NTHREADS, 0, stream>>>(
        x, W_ih, W_hh, b_ih, b_hh, W_out, b_out, y);
}
